// Mamba_Layer_25503515803681
// MI455X (gfx1250) — matrix operand order repaired and checked by reading, not compiled, not run on hardware
//
#include <hip/hip_runtime.h>
#include <math.h>

// ---------------------------------------------------------------------------
// Mamba 2-layer block for MI455X (gfx1250, wave32, WMMA).
// Projections: bf16 WMMA GEMM (v_wmma_f32_16x16x32_bf16, fp32 accum) with
// double-buffered LDS staged via GLOBAL_LOAD_ASYNC_TO_LDS_B128 (ASYNCcnt).
// ---------------------------------------------------------------------------

typedef __bf16 bf16;
typedef __attribute__((ext_vector_type(16))) __bf16 v16bf;
typedef __attribute__((ext_vector_type(8)))  float  v8f;

#define NB      4
#define SEQ     1024
#define DMODEL  1024
#define DINNER  2048
#define DSTATE  16
#define DTRANK  64
#define NROWS   (NB * SEQ)       // 4096
#define XZW     (2 * DINNER)     // 4096

__device__ __forceinline__ float siluf(float x) { return x / (1.0f + __expf(-x)); }
__device__ __forceinline__ float softplusf_(float x) { return x > 20.0f ? x : log1pf(__expf(x)); }

// Async DMA: 16 bytes global -> LDS per lane, tracked by ASYNCcnt (no VGPR
// round trip). LDS offset = low 32 bits of the generic shared address.
__device__ __forceinline__ void async_b128(unsigned lds_off, const bf16* gptr) {
    asm volatile("global_load_async_to_lds_b128 %0, %1, off"
                 :: "v"(lds_off), "v"(gptr) : "memory");
}
__device__ __forceinline__ void wait_async0() {
    asm volatile("s_wait_asynccnt 0x0" ::: "memory");
}

// ---------------------------------------------------------------------------
// bf16 WMMA GEMM: C[M,N] = A[M,K] * W[N,K]^T (+bias, optional softplus).
// Block tile 128(M) x TBN(N), K-chunk 32, double-buffered async LDS staging.
// 8 waves: 4 in M x 2 in N. TBN=64: wave tile 32x32 = 4 WMMA/chunk;
// TBN=32: wave tile 32x16 = 2 WMMA/chunk (for N=96).
// ---------------------------------------------------------------------------
#define BM 128
#define BK 32
#define PADK 8   // keeps rows 80B apart: 16B aligned, banks spread

template <int TBN>
__global__ __launch_bounds__(256) void gemm_bf16_wmma(
    const bf16* __restrict__ A, const bf16* __restrict__ W,
    float* __restrict__ C, const float* __restrict__ bias,
    int M, int N, int K, int act)
{
    constexpr int NFRAG = TBN / 32;  // B fragments per wave (64->2, 32->1)
    __shared__ __align__(16) bf16 sA[2][BM][BK + PADK];
    __shared__ __align__(16) bf16 sW[2][TBN][BK + PADK];

    const int tid  = threadIdx.x;
    const int lane = tid & 31;
    const int wave = tid >> 5;
    const int wm   = wave & 3;   // wave row (32 rows each)
    const int wn   = wave >> 2;  // wave col
    const int m0   = blockIdx.y * BM;
    const int n0   = blockIdx.x * TBN;

    v8f zero = {};
    v8f acc[NFRAG][2];
#pragma unroll
    for (int f = 0; f < NFRAG; ++f) { acc[f][0] = zero; acc[f][1] = zero; }

    // A tile: 128x32 bf16 = 512 x 16B chunks -> 2 per thread.
    const int ar = tid >> 1;
    const int ac = (tid & 1) * 16;
    // W tile: TBN x 32 bf16 = TBN*4 x 16B chunks -> 1 per thread (guarded).
    const int wrow = tid >> 2;
    const int wseg = (tid & 3) * 8;

    auto stage = [&](int k0, int buf) {
        const bf16* asrc = A + (size_t)(m0 + ar) * K + (k0 + ac);
        async_b128((unsigned)(size_t)&sA[buf][ar][ac],     asrc);
        async_b128((unsigned)(size_t)&sA[buf][ar][ac + 8], asrc + 8);
        if (TBN == 64 || tid < 128) {
            const bf16* wsrc = W + (size_t)(n0 + wrow) * K + (k0 + wseg);
            async_b128((unsigned)(size_t)&sW[buf][wrow][wseg], wsrc);
        }
    };

    stage(0, 0);
    wait_async0();
    __syncthreads();

    int buf = 0;
    for (int k0 = 0; k0 < K; k0 += BK) {
        if (k0 + BK < K) stage(k0 + BK, buf ^ 1);   // prefetch next chunk

        // A fragments (16x32): lanes 0-15 hold K {0..7,16..23}, lanes 16-31
        // hold K {8..15,24..31} (ISA 7.12.2 16-bit A layout).
        const int arow = wm * 32 + (lane & 15);
        const int kba  = (lane >> 4) * 8;
        v16bf af0, af1;
#pragma unroll
        for (int e = 0; e < 16; ++e) {
            const int k = kba + (e & 7) + ((e >> 3) << 4);
            af0[e] = sA[buf][arow][k];
            af1[e] = sA[buf][arow + 16][k];
        }
#pragma unroll
        for (int f = 0; f < NFRAG; ++f) {
            // B fragment (32x16): lane&15 = column, same K order as A.
            const int col = wn * (16 * NFRAG) + f * 16 + (lane & 15);
            const int kb  = (lane >> 4) * 8;
            v16bf bfrag;
#pragma unroll
            for (int e = 0; e < 16; ++e) bfrag[e] = sW[buf][col][kb + (e & 7) + ((e >> 3) << 4)];
            acc[f][0] = __builtin_amdgcn_wmma_f32_16x16x32_bf16(
                false, af0, false, bfrag, (short)0, acc[f][0], false, false);
            acc[f][1] = __builtin_amdgcn_wmma_f32_16x16x32_bf16(
                false, af1, false, bfrag, (short)0, acc[f][1], false, false);
        }
        wait_async0();      // this wave's prefetch landed in LDS
        __syncthreads();    // everyone done reading buf / writing buf^1
        buf ^= 1;
    }

    // Epilogue. C/D layout: VGPR r -> M=r (lanes 0-15) / M=r+8 (lanes 16-31).
    const int mb = m0 + wm * 32 + ((lane >> 4) << 3);
#pragma unroll
    for (int f = 0; f < NFRAG; ++f) {
        const int ncol = n0 + wn * (16 * NFRAG) + f * 16 + (lane & 15);
        const float bv = bias ? bias[ncol] : 0.0f;
#pragma unroll
        for (int r = 0; r < 8; ++r) {
            float v0 = acc[f][0][r] + bv;
            float v1 = acc[f][1][r] + bv;
            if (act == 1) { v0 = softplusf_(v0); v1 = softplusf_(v1); }
            C[(size_t)(mb + r) * N + ncol]      = v0;
            C[(size_t)(mb + r + 16) * N + ncol] = v1;
        }
    }
}

// ---------------------------------------------------------------------------
// Residual add + RMSNorm. One block per row of D_MODEL=1024.
// ---------------------------------------------------------------------------
__global__ __launch_bounds__(256) void addnorm_kernel(
    const float* __restrict__ x, const float* __restrict__ resin,
    const float* __restrict__ w,
    float* __restrict__ resout, bf16* __restrict__ outb, float* __restrict__ outf)
{
    const int row = blockIdx.x;
    const size_t base = (size_t)row * DMODEL;
    float v[4];
    float ss = 0.0f;
#pragma unroll
    for (int i = 0; i < 4; ++i) {
        const int c = threadIdx.x + i * 256;
        float t = x[base + c];
        if (resin) t += resin[base + c];
        v[i] = t;
        ss += t * t;
        if (resout) resout[base + c] = t;
    }
#pragma unroll
    for (int o = 16; o > 0; o >>= 1) ss += __shfl_xor(ss, o, 32);
    __shared__ float red[8];
    if ((threadIdx.x & 31) == 0) red[threadIdx.x >> 5] = ss;
    __syncthreads();
    float tot = 0.0f;
#pragma unroll
    for (int i = 0; i < 8; ++i) tot += red[i];
    const float scale = rsqrtf(tot * (1.0f / DMODEL) + 1e-5f);
#pragma unroll
    for (int i = 0; i < 4; ++i) {
        const int c = threadIdx.x + i * 256;
        const float o = v[i] * scale * w[c];
        if (outb) outb[base + c] = (bf16)o;
        if (outf) outf[base + c] = o;
    }
}

// ---------------------------------------------------------------------------
// Depthwise causal conv (D_CONV=4) + bias + SiLU over xi = xz[:, :DINNER].
// ---------------------------------------------------------------------------
__global__ __launch_bounds__(256) void conv_silu_kernel(
    const float* __restrict__ xz, const float* __restrict__ cw,
    const float* __restrict__ cb,
    float* __restrict__ xcf, bf16* __restrict__ xcb)
{
    const size_t idx = (size_t)blockIdx.x * 256 + threadIdx.x;  // B*L*DINNER
    const int d = (int)(idx & (DINNER - 1));
    const size_t bl = idx >> 11;                 // row index b*SEQ + l
    const int l = (int)(bl & (SEQ - 1));
    const size_t brow = bl - (size_t)l;          // b*SEQ
    float acc = cb[d];
#pragma unroll
    for (int t = 0; t < 4; ++t) {
        const int ls = l - 3 + t;
        if (ls >= 0)
            acc += cw[d * 4 + t] * xz[(brow + (size_t)ls) * XZW + d];
    }
    const float s = siluf(acc);
    xcf[idx] = s;
    xcb[idx] = (bf16)s;
}

// dt_r = bf16(x_dbl[:, :64])
__global__ __launch_bounds__(256) void extract_dtr_kernel(
    const float* __restrict__ xdbl, bf16* __restrict__ dtr)
{
    const int idx = blockIdx.x * 256 + threadIdx.x;   // NROWS*64
    const int c = idx & 63;
    const int row = idx >> 6;
    dtr[idx] = (bf16)xdbl[(size_t)row * 96 + c];
}

// ---------------------------------------------------------------------------
// Selective scan: one thread per (batch, channel). h[16] lives in VGPRs,
// B_t/C_t (shared across channels) staged through LDS in 64-step chunks.
// ---------------------------------------------------------------------------
#define CHUNK 64
__global__ __launch_bounds__(256) void scan_kernel(
    const float* __restrict__ dt, const float* __restrict__ xc,
    const float* __restrict__ xdbl, const float* __restrict__ A_log,
    float* __restrict__ y)
{
    const int b = blockIdx.x >> 3;                       // DINNER/256 = 8 blocks/batch
    const int d = (blockIdx.x & 7) * 256 + threadIdx.x;  // channel
    float Arow[DSTATE];
#pragma unroll
    for (int s = 0; s < DSTATE; ++s) Arow[s] = -__expf(A_log[d * DSTATE + s]);
    float h[DSTATE];
#pragma unroll
    for (int s = 0; s < DSTATE; ++s) h[s] = 0.0f;

    __shared__ float sB[CHUNK][DSTATE];
    __shared__ float sC[CHUNK][DSTATE];

    for (int lc = 0; lc < SEQ; lc += CHUNK) {
        __syncthreads();
        for (int i = threadIdx.x; i < CHUNK * DSTATE; i += 256) {
            const int ll = i >> 4, s = i & 15;
            const size_t r = (size_t)(b * SEQ + lc + ll) * 96;
            sB[ll][s] = xdbl[r + DTRANK + s];
            sC[ll][s] = xdbl[r + DTRANK + DSTATE + s];
        }
        __syncthreads();
        for (int ll = 0; ll < CHUNK; ++ll) {
            const size_t off = (size_t)(b * SEQ + lc + ll) * DINNER + d;
            const float dtv = dt[off];
            const float dx  = dtv * xc[off];
            float acc = 0.0f;
#pragma unroll
            for (int s = 0; s < DSTATE; ++s) {
                h[s] = __expf(dtv * Arow[s]) * h[s] + dx * sB[ll][s];
                acc += h[s] * sC[ll][s];
            }
            y[off] = acc;
        }
    }
}

// y2 = bf16((y + xc*D) * silu(z))
__global__ __launch_bounds__(256) void gate_kernel(
    const float* __restrict__ y, const float* __restrict__ xc,
    const float* __restrict__ xz, const float* __restrict__ Dsk,
    bf16* __restrict__ out)
{
    const size_t idx = (size_t)blockIdx.x * 256 + threadIdx.x;
    const int d = (int)(idx & (DINNER - 1));
    const size_t row = idx >> 11;
    const float z = xz[row * XZW + DINNER + d];
    const float val = (y[idx] + xc[idx] * Dsk[d]) * siluf(z);
    out[idx] = (bf16)val;
}

// fp32 -> bf16 weight conversion
__global__ __launch_bounds__(256) void f2b_kernel(
    const float* __restrict__ src, bf16* __restrict__ dst, int n)
{
    const int i = blockIdx.x * 256 + threadIdx.x;
    if (i < n) dst[i] = (bf16)src[i];
}

// ---------------------------------------------------------------------------
// Launcher
// ---------------------------------------------------------------------------
extern "C" void kernel_launch(void* const* d_in, const int* in_sizes, int n_in,
                              void* d_out, int out_size, void* d_ws, size_t ws_size,
                              hipStream_t stream)
{
    (void)in_sizes; (void)n_in; (void)out_size; (void)ws_size;

    const float* hidden_states = (const float*)d_in[0];
    const float* norm_w     = (const float*)d_in[1];
    const float* in_proj_w  = (const float*)d_in[2];
    const float* conv_w     = (const float*)d_in[3];
    const float* conv_b     = (const float*)d_in[4];
    const float* x_proj_w   = (const float*)d_in[5];
    const float* dt_proj_w  = (const float*)d_in[6];
    const float* dt_proj_b  = (const float*)d_in[7];
    const float* A_log      = (const float*)d_in[8];
    const float* Dskip      = (const float*)d_in[9];
    const float* out_proj_w = (const float*)d_in[10];
    const float* norm_f_w   = (const float*)d_in[11];

    char* p = (char*)d_ws;
    auto take = [&](size_t bytes) -> void* {
        void* r = (void*)p;
        p += (bytes + 255) & ~(size_t)255;
        return r;
    };

    float* residual = (float*)take(sizeof(float) * (size_t)NROWS * DMODEL);
    bf16*  normed   = (bf16*) take(sizeof(bf16)  * (size_t)NROWS * DMODEL);
    float* xz       = (float*)take(sizeof(float) * (size_t)NROWS * XZW);
    float* xcf      = (float*)take(sizeof(float) * (size_t)NROWS * DINNER);
    bf16*  xcb      = (bf16*) take(sizeof(bf16)  * (size_t)NROWS * DINNER);
    float* xdbl     = (float*)take(sizeof(float) * (size_t)NROWS * 96);
    bf16*  dtr      = (bf16*) take(sizeof(bf16)  * (size_t)NROWS * DTRANK);
    float* dtbuf    = (float*)take(sizeof(float) * (size_t)NROWS * DINNER);
    float* ybuf     = (float*)take(sizeof(float) * (size_t)NROWS * DINNER);
    bf16*  ygate    = (bf16*) take(sizeof(bf16)  * (size_t)NROWS * DINNER);
    float* mixout   = (float*)take(sizeof(float) * (size_t)NROWS * DMODEL);
    bf16*  w_in_b   = (bf16*) take(sizeof(bf16)  * 2ull * XZW * DMODEL);
    bf16*  w_x_b    = (bf16*) take(sizeof(bf16)  * 2ull * 96 * DINNER);
    bf16*  w_dt_b   = (bf16*) take(sizeof(bf16)  * 2ull * DINNER * DTRANK);
    bf16*  w_out_b  = (bf16*) take(sizeof(bf16)  * 2ull * DMODEL * DINNER);

    auto cvt = [&](const float* s, bf16* dst, int n) {
        f2b_kernel<<<(n + 255) / 256, 256, 0, stream>>>(s, dst, n);
    };
    cvt(in_proj_w,  w_in_b,  2 * XZW * DMODEL);
    cvt(x_proj_w,   w_x_b,   2 * 96 * DINNER);
    cvt(dt_proj_w,  w_dt_b,  2 * DINNER * DTRANK);
    cvt(out_proj_w, w_out_b, 2 * DMODEL * DINNER);

    const float* cur_hidden = hidden_states;
    for (int i = 0; i < 2; ++i) {
        // residual += hidden; normed = rmsnorm(residual) * norm_w[i]  (bf16)
        addnorm_kernel<<<NROWS, 256, 0, stream>>>(
            cur_hidden, i == 0 ? nullptr : residual, norm_w + i * DMODEL,
            residual, normed, nullptr);

        // xz = normed @ in_proj_w^T   [4096 x 4096]
        gemm_bf16_wmma<64><<<dim3(XZW / 64, NROWS / BM), 256, 0, stream>>>(
            normed, w_in_b + (size_t)i * XZW * DMODEL, xz, nullptr,
            NROWS, XZW, DMODEL, 0);

        // xc = silu(depthwise_conv(xi) + b)
        conv_silu_kernel<<<(NROWS * DINNER) / 256, 256, 0, stream>>>(
            xz, conv_w + i * DINNER * 4, conv_b + i * DINNER, xcf, xcb);

        // x_dbl = xc @ x_proj_w^T   [4096 x 96]
        gemm_bf16_wmma<32><<<dim3(96 / 32, NROWS / BM), 256, 0, stream>>>(
            xcb, w_x_b + (size_t)i * 96 * DINNER, xdbl, nullptr,
            NROWS, 96, DINNER, 0);

        extract_dtr_kernel<<<(NROWS * DTRANK) / 256, 256, 0, stream>>>(xdbl, dtr);

        // dt = softplus(dt_r @ dt_proj_w^T + dt_proj_b)   [4096 x 2048]
        gemm_bf16_wmma<64><<<dim3(DINNER / 64, NROWS / BM), 256, 0, stream>>>(
            dtr, w_dt_b + (size_t)i * DINNER * DTRANK, dtbuf,
            dt_proj_b + i * DINNER, NROWS, DINNER, DTRANK, 1);

        // selective scan
        scan_kernel<<<NB * (DINNER / 256), 256, 0, stream>>>(
            dtbuf, xcf, xdbl, A_log + (size_t)i * DINNER * DSTATE, ybuf);

        // gate: y2 = (y + xc*D) * silu(z)
        gate_kernel<<<(NROWS * DINNER) / 256, 256, 0, stream>>>(
            ybuf, xcf, xz, Dskip + i * DINNER, ygate);

        // hidden = y2 @ out_proj_w^T   [4096 x 1024]
        gemm_bf16_wmma<64><<<dim3(DMODEL / 64, NROWS / BM), 256, 0, stream>>>(
            ygate, w_out_b + (size_t)i * DMODEL * DINNER, mixout, nullptr,
            NROWS, DMODEL, DINNER, 0);

        cur_hidden = mixout;
    }

    // out = rmsnorm(hidden + residual, norm_f_w)  (fp32)
    addnorm_kernel<<<NROWS, 256, 0, stream>>>(
        mixout, residual, norm_f_w, nullptr, nullptr, (float*)d_out);
}
